// GraphConvolution_59588376264936
// MI455X (gfx1250) — matrix operand order repaired and checked by reading, not compiled, not run on hardware
//
#include <hip/hip_runtime.h>
#include <hip/hip_bf16.h>

#define N_NODES 8192
#define F_IN    256
#define F_OUT   128
#define KSPLIT  8
#define NCHUNKS (N_NODES / 32)            // 256 K-chunks of 32
#define SCHUNKS (NCHUNKS / KSPLIT)        // 32 chunks per K-slice

typedef __attribute__((ext_vector_type(16))) __bf16       v16bf;
typedef __attribute__((ext_vector_type(8)))  float        v8f;
typedef __attribute__((ext_vector_type(8)))  unsigned int v8u;
typedef __attribute__((ext_vector_type(2)))  float        v2f;

union Frag {
    v8u          u8;
    unsigned int u[8];
    v16bf        v;
};

// round-to-nearest-even f32 -> bf16, packed pair (a in lo, b in hi)
__device__ __forceinline__ unsigned int pack_bf16x2(float a, float b) {
    unsigned int ua = __float_as_uint(a);
    unsigned int ub = __float_as_uint(b);
    ua += 0x7FFFu + ((ua >> 16) & 1u);
    ub += 0x7FFFu + ((ub >> 16) & 1u);
    return (ua >> 16) | (ub & 0xFFFF0000u);
}

// ---------------------------------------------------------------------------
// Kernel 1: support = input @ weight (f32). One block per row, 128 threads =
// 128 columns; input row in LDS, weight (128 KB) L2-resident.
// ---------------------------------------------------------------------------
__global__ void gcn_support_kernel(const float* __restrict__ input,
                                   const float* __restrict__ weight,
                                   float* __restrict__ support) {
    __shared__ float xrow[F_IN];
    const int row = blockIdx.x;
    const int col = threadIdx.x;
    for (int i = threadIdx.x; i < F_IN; i += 128)
        xrow[i] = input[(size_t)row * F_IN + i];
    __syncthreads();
    float acc = 0.0f;
#pragma unroll 8
    for (int k = 0; k < F_IN; ++k)
        acc = fmaf(xrow[k], weight[(size_t)k * F_OUT + col], acc);
    support[(size_t)row * F_OUT + col] = acc;
}

// ---------------------------------------------------------------------------
// Kernel 2: repack support f32 [K=8192][N=128] into bf16 B fragments for
// v_wmma_f32_16x16x32_bf16 (ISA 7.12.2: VGPR v, K=(v>=4?16:0)+(lane/16)*8
// +(v%4)*2, pair; N = lane%16). Dword index ((c*8+t)*32+lane)*8+v so
// each lane's 8 dwords are contiguous -> two global_load_b128 in the GEMM.
// ---------------------------------------------------------------------------
__global__ void gcn_repack_kernel(const float* __restrict__ support,
                                  unsigned int* __restrict__ bfrag) {
    const int idx = blockIdx.x * blockDim.x + threadIdx.x; // 0 .. 524287
    const int v    = idx & 7;
    const int lane = (idx >> 3) & 31;
    const int t    = (idx >> 8) & 7;
    const int c    = idx >> 11;
    const int kb   = ((v & 4) << 2) + ((lane >> 4) << 3) + 2 * (v & 3);
    const int n    = t * 16 + (lane & 15);
    const int k    = c * 32 + kb;
    const float a  = support[(size_t)k * F_OUT + n];
    const float b  = support[(size_t)(k + 1) * F_OUT + n];
    bfrag[idx] = pack_bf16x2(a, b);
}

// ---------------------------------------------------------------------------
// Kernel 3: K-split partial SpMM. blockIdx.x = row tile (512), blockIdx.y =
// K-slice (8) -> 4096 independent waves for latency hiding. Each wave streams
// its 16-row x 1024-K slab of adj once (non-temporal), folds +I into the
// bf16 A fragment, accumulates partial row-sums, and runs 8 WMMAs per chunk
// against the pre-swizzled, L2-resident B fragments.
// A layout (ISA 7.12.2, 16-bit 16x32): row M = lane%16; VGPR v ->
// K = (v>=4 ? 16:0) + (lane/16)*8 + (v%4)*2, packed pair.
// ---------------------------------------------------------------------------
__global__ void gcn_spmm_partial(const float* __restrict__ adj,
                                 const unsigned int* __restrict__ bfrag,
                                 float* __restrict__ partial,
                                 float* __restrict__ rowsum_part) {
    const int lane = threadIdx.x;          // 0..31
    const int r    = lane & 15;
    const int half = lane >> 4;
    const int tile = blockIdx.x;           // 0..511
    const int ks   = blockIdx.y;           // 0..7
    const int row0 = tile * 16;
    const int grow = row0 + r;
    const int c0   = ks * SCHUNKS;
    const float* __restrict__ arow = adj + (size_t)grow * N_NODES;
    const v8u* __restrict__ bfv = (const v8u*)bfrag;

    v8f acc[8];
#pragma unroll
    for (int t = 0; t < 8; ++t) acc[t] = (v8f)(0.0f);

    float rowsum = 0.0f;

    // per-lane K offsets of this lane's A-fragment pairs within a chunk
    int koff[8];
#pragma unroll
    for (int v = 0; v < 8; ++v)
        koff[v] = ((v & 4) << 2) + half * 8 + (v & 3) * 2;

    // prologue: raw A f32 for first chunk of this slice
    v2f araw[8];
#pragma unroll
    for (int v = 0; v < 8; ++v)
        araw[v] = __builtin_nontemporal_load((const v2f*)(arow + c0 * 32 + koff[v]));

    for (int cc = 0; cc < SCHUNKS; ++cc) {
        const int c     = c0 + cc;
        const int kbase = c * 32;

        // ---- B fragments for this chunk (L2-resident) ----
        Frag B[8];
        const size_t bbase = (size_t)c * 256 + lane;
#pragma unroll
        for (int t = 0; t < 8; ++t)
            B[t].u8 = bfv[bbase + (size_t)t * 32];

        // ---- raw A for next chunk (wraps inside slice: always in-bounds) --
        const int nbase = (c0 + ((cc + 1) & (SCHUNKS - 1))) * 32;
        v2f anext[8];
#pragma unroll
        for (int v = 0; v < 8; ++v)
            anext[v] = __builtin_nontemporal_load((const v2f*)(arow + nbase + koff[v]));

        // ---- convert A to bf16 fragment (+I diagonal), degree partial ----
        Frag A;
        float asum = 0.0f;
#pragma unroll
        for (int v = 0; v < 8; ++v) {
            const int k = kbase + koff[v];
            float x0 = araw[v][0];
            float x1 = araw[v][1];
            if (grow == k)     x0 += 1.0f;   // adj + I (once, owning slice)
            if (grow == k + 1) x1 += 1.0f;
            asum += x0 + x1;
            A.u[v] = pack_bf16x2(x0, x1);
        }
        rowsum += asum;

        // ---- 8 WMMAs: 16 rows x 128 cols, K += 32 ----
#pragma unroll
        for (int t = 0; t < 8; ++t)
            acc[t] = __builtin_amdgcn_wmma_f32_16x16x32_bf16(
                false, A.v, false, B[t].v, (short)0, acc[t], false, false);

#pragma unroll
        for (int v = 0; v < 8; ++v) araw[v] = anext[v];
    }

    // combine the two half-lane partial degree sums for each row
    rowsum += __shfl_xor(rowsum, 16, 32);   // lane x holds row (x&15)

    // write partials: partial[ks][row][col], rowsum_part[ks][row]
    float* __restrict__ pslab = partial + (size_t)ks * N_NODES * F_OUT;
#pragma unroll
    for (int j = 0; j < 8; ++j) {
        const int m = j + half * 8;                    // D: VGPR j, halves
        const size_t orow = (size_t)(row0 + m) * F_OUT;
#pragma unroll
        for (int t = 0; t < 8; ++t)
            pslab[orow + t * 16 + r] = acc[t][j];
    }
    if (lane < 16)
        rowsum_part[ks * N_NODES + grow] = rowsum;
}

// ---------------------------------------------------------------------------
// Kernel 4: reduce K-slices, apply deg_inv = 1/(1+sum) and bias.
// ---------------------------------------------------------------------------
__global__ void gcn_finalize_kernel(const float* __restrict__ partial,
                                    const float* __restrict__ rowsum_part,
                                    const float* __restrict__ bias,
                                    float* __restrict__ out) {
    const int idx = blockIdx.x * blockDim.x + threadIdx.x; // 0 .. 1048575
    const int col = idx & (F_OUT - 1);
    const int row = idx >> 7;
    float s  = 0.0f;
    float rs = 1.0f;                       // +1: self-loop degree
#pragma unroll
    for (int ks = 0; ks < KSPLIT; ++ks) {
        s  += partial[(size_t)ks * N_NODES * F_OUT + idx];
        rs += rowsum_part[ks * N_NODES + row];
    }
    out[idx] = s * (1.0f / rs) + bias[col];
}

extern "C" void kernel_launch(void* const* d_in, const int* in_sizes, int n_in,
                              void* d_out, int out_size, void* d_ws, size_t ws_size,
                              hipStream_t stream) {
    const float* input  = (const float*)d_in[0];   // [8192,256]
    const float* adj    = (const float*)d_in[1];   // [8192,8192]
    const float* weight = (const float*)d_in[2];   // [256,128]
    const float* bias   = (const float*)d_in[3];   // [128]
    float* out = (float*)d_out;                    // [8192,128]

    char* ws = (char*)d_ws;
    float*        support     = (float*)ws;                          // 4 MB
    unsigned int* bfrag       = (unsigned int*)(ws + (4u << 20));    // 2 MB
    float*        partial     = (float*)(ws + (6u << 20));           // 32 MB
    float*        rowsum_part = (float*)(ws + (38u << 20));          // 256 KB

    // 1) support = input @ weight (f32)
    gcn_support_kernel<<<N_NODES, 128, 0, stream>>>(input, weight, support);

    // 2) repack support into bf16 WMMA B fragments
    const int total = NCHUNKS * 8 * 32 * 8;          // 524288 dwords
    gcn_repack_kernel<<<total / 256, 256, 0, stream>>>(support, bfrag);

    // 3) K-split partial SpMM via v_wmma_f32_16x16x32_bf16 (4096 waves)
    gcn_spmm_partial<<<dim3(N_NODES / 16, KSPLIT), 32, 0, stream>>>(
        adj, bfrag, partial, rowsum_part);

    // 4) reduce slices, normalize, bias
    gcn_finalize_kernel<<<(N_NODES * F_OUT) / 256, 256, 0, stream>>>(
        partial, rowsum_part, bias, out);
}
